// BaseEncoder_39436389712076
// MI455X (gfx1250) — compile-verified, matrix operand order checked by reading, not run on hardware
//
#include <hip/hip_runtime.h>

// ---------------------------------------------------------------------------
// BiLSTM encoder for MI455X (gfx1250, wave32, WMMA).
// Precision: bf16 inputs to v_wmma_f32_16x16x32_bf16, f32 accumulate/state.
// Workspace layout (needs ~176MB):
//   [0,1MB)    S table   (char-vocab x K x NF precomputed conv dot products)
//   [1,7MB)    packed W_ih (6 x 1MB bf16 fragments)
//   [7,10MB)   packed W_hh (6 x 0.5MB bf16 fragments)
//   [16,32MB)  x ping (bf16, BT x 512)
//   [32,48MB)  x pong (bf16, BT x 512)
//   [48,176MB) gx (f32, 2 dirs x BT x 4H)
// ---------------------------------------------------------------------------

typedef __attribute__((ext_vector_type(16))) __bf16 v16bf;
typedef __attribute__((ext_vector_type(8)))  __bf16 v8bf;
typedef __attribute__((ext_vector_type(8)))  float  v8f;

#define B_   64
#define T_   256
#define L_   20
#define WD_  300
#define CD_  100
#define PD_  100
#define NF_  100
#define KC_  3
#define H_   256
#define DP_  512           // padded input width (D0=500 -> 512, D1=512)
#define G4_  1024          // 4*H
#define BT_  (B_*T_)
#define CV_  100

// A-fragment load: lane holds K = kbase+0..7 and kbase+16..23 (two 16B chunks)
__device__ __forceinline__ v16bf ld_fragA(const __bf16* p) {
  v8bf lo = *(const v8bf*)(p);
  v8bf hi = *(const v8bf*)(p + 16);
  return __builtin_shufflevector(lo, hi, 0,1,2,3,4,5,6,7,8,9,10,11,12,13,14,15);
}
// Packed-fragment load: 16 contiguous bf16 (32B) per lane
__device__ __forceinline__ v16bf ld_fragP(const __bf16* p) {
  return *(const v16bf*)p;
}
__device__ __forceinline__ float sigmoidf_(float x) {
  return 1.0f / (1.0f + __expf(-x));
}

// ---------------------------------------------------------------------------
// S[v][k][f] = sum_c emb_char[v][c] * conv_w[f][c][k]; row v==CV_ is zeros
// (zero-padding sentinel for out-of-range char positions).
// ---------------------------------------------------------------------------
__global__ void build_S_kernel(const float* __restrict__ emb_char,
                               const float* __restrict__ conv_w,
                               float* __restrict__ S) {
  int i = blockIdx.x * blockDim.x + threadIdx.x;
  const int total = (CV_ + 1) * KC_ * NF_;
  if (i >= total) return;
  int f = i % NF_;
  int k = (i / NF_) % KC_;
  int v = i / (NF_ * KC_);
  float s = 0.f;
  if (v < CV_) {
    for (int c = 0; c < CD_; ++c)
      s += emb_char[v * CD_ + c] * conv_w[(f * CD_ + c) * KC_ + k];
  }
  S[(v * KC_ + k) * NF_ + f] = s;
}

// ---------------------------------------------------------------------------
// Pack weight (N_total=1024 rows x Kreal cols, row-major, B[k][n]=w[n][k])
// into WMMA bf16 B-fragment layout: block (n_tile,kc), lane l, elem e ->
//   n = n_tile*16 + l%16,  k = kc*32 + (l/16)*8 + (e%8) + (e/8)*16
// 16 contiguous bf16 per lane. k >= Kreal zero-filled.
// ---------------------------------------------------------------------------
__global__ void pack_w_kernel(const float* __restrict__ w, int Kreal, int kchunks,
                              __bf16* __restrict__ out) {
  int bi   = blockIdx.x;              // n_tile*kchunks + kc
  int lane = threadIdx.x;             // 0..31
  int n_tile = bi / kchunks;
  int kc     = bi % kchunks;
  int n      = n_tile * 16 + (lane & 15);
  int kbase  = kc * 32 + ((lane >> 4) << 3);
  __bf16* o = out + ((size_t)bi * 32 + lane) * 16;
  for (int e = 0; e < 16; ++e) {
    int k = kbase + (e & 7) + ((e >> 3) << 4);
    float v = (k < Kreal) ? w[(size_t)n * Kreal + k] : 0.0f;
    o[e] = (__bf16)v;
  }
}

// ---------------------------------------------------------------------------
// Embedding gather + char conv-max-tanh + concat -> x0 bf16 (BT x 512),
// cols: [0,300)=word  [300,400)=char  [400,500)=pos  [500,512)=zero pad.
// ---------------------------------------------------------------------------
#define TOK_PER_BLK 128
__global__ void embed_kernel(const int* __restrict__ iw, const int* __restrict__ ic,
                             const int* __restrict__ ip,
                             const float* __restrict__ emb_word,
                             const float* __restrict__ emb_pos,
                             const float* __restrict__ S,
                             const float* __restrict__ conv_b,
                             __bf16* __restrict__ xb) {
  __shared__ float Sl[(CV_ + 1) * KC_ * NF_];   // ~121KB of 320KB LDS
  __shared__ int   cid[L_ + 4];                 // char id per padded position
  int tid = threadIdx.x;                        // 128 threads
  for (int i = tid; i < (CV_ + 1) * KC_ * NF_; i += 128) Sl[i] = S[i];
  __syncthreads();
  for (int it = 0; it < TOK_PER_BLK; ++it) {
    int tk = blockIdx.x * TOK_PER_BLK + it;     // flattened b*T + t
    if (tid < L_ + 4) {
      int q = tid - 2;                          // q = position index -2..21
      cid[tid] = (q >= 0 && q < L_) ? ic[tk * L_ + q] : CV_;
    }
    __syncthreads();
    __bf16* xrow = xb + (size_t)tk * DP_;
    int wv = iw[tk];
    for (int i = tid; i < WD_; i += 128)
      xrow[i] = (__bf16)emb_word[(size_t)wv * WD_ + i];
    int pv = ip[tk];
    for (int i = tid; i < PD_; i += 128)
      xrow[WD_ + NF_ + i] = (__bf16)emb_pos[pv * PD_ + i];
    if (tid < DP_ - (WD_ + NF_ + PD_))
      xrow[WD_ + NF_ + PD_ + tid] = (__bf16)0.0f;
    if (tid < NF_) {
      float best = -1e30f;
      for (int p = 0; p < L_ + KC_ - 1; ++p) {  // 22 conv output positions
        float s = Sl[(cid[p]     * KC_ + 0) * NF_ + tid]
                + Sl[(cid[p + 1] * KC_ + 1) * NF_ + tid]
                + Sl[(cid[p + 2] * KC_ + 2) * NF_ + tid];
        best = fmaxf(best, s);
      }
      xrow[WD_ + tid] = (__bf16)tanhf(best + conv_b[tid]);
    }
    __syncthreads();
  }
}

// ---------------------------------------------------------------------------
// gx[dir] = x(bf16, BT x 512) @ W_ih[dir]^T + bias[dir]  -> f32 (2,BT,1024)
// One 16x16 output tile per wave; K=512 in 16 chunks of 32 via bf16 WMMA.
// Grid covers 2 dirs x 1024 M-tiles x 64 N-tiles = 131072 tiles, 8 waves/blk.
// ---------------------------------------------------------------------------
__global__ void __launch_bounds__(256) gemm_gx_kernel(
    const __bf16* __restrict__ x,
    const __bf16* __restrict__ wpack,   // both dirs, packed
    const float*  __restrict__ bias,    // (2, 4H)
    float*        __restrict__ gx) {    // (2, BT, 4H)
  int lane = threadIdx.x & 31;
  int gid  = blockIdx.x * 8 + (threadIdx.x >> 5);
  int dir  = gid >> 16;
  int rem  = gid & 0xFFFF;
  int m_tile = rem >> 6;
  int n_tile = rem & 63;
  const __bf16* wp = wpack + (size_t)dir * (64 * 16 * 32 * 16);
  float bv = bias[dir * G4_ + n_tile * 16 + (lane & 15)];
  v8f c;
  for (int e = 0; e < 8; ++e) c[e] = bv;
  const __bf16* xrow =
      x + (size_t)(m_tile * 16 + (lane & 15)) * DP_ + ((lane >> 4) << 3);
  for (int kc = 0; kc < 16; ++kc) {
    v16bf a = ld_fragA(xrow + kc * 32);
    v16bf b = ld_fragP(wp + ((size_t)(n_tile * 16 + kc) * 32 + lane) * 16);
    c = __builtin_amdgcn_wmma_f32_16x16x32_bf16(false, a, false, b,
                                                (short)0, c, false, false);
  }
  float* gcol = gx + (size_t)dir * BT_ * G4_ + (n_tile * 16 + (lane & 15));
  int mbase = m_tile * 16 + ((lane >> 4) << 3);
  for (int e = 0; e < 8; ++e)
    gcol[(size_t)(mbase + e) * G4_] = c[e];
}

// ---------------------------------------------------------------------------
// LSTM scan. 8 blocks = 4 batch-groups (16 rows) x 2 dirs, 512 thr = 16 waves.
// Per step: gates(16x1024) = gx[:,t,:] + h_prev(16x256) @ W_hh^T via WMMA.
// Wave w owns gate tiles {w, w+16, w+32, w+48} = (i,f,g,o) for hidden cols
// w*16..w*16+15, so the pointwise update stays lane-local. h shared via LDS.
// ---------------------------------------------------------------------------
__global__ void __launch_bounds__(512) lstm_scan_kernel(
    const float*  __restrict__ gx,     // (2, BT, 4H)
    const __bf16* __restrict__ whh,    // (2, packed 64x8 blocks)
    const float*  __restrict__ mask,   // (B, T)
    float*        __restrict__ outf,   // (B, T, 512) f32, or nullptr
    __bf16*       __restrict__ outb) { // (B, T, 512) bf16 (next layer input)
  __shared__ __align__(32) __bf16 hbuf[16 * H_];   // 8KB
  int tid  = threadIdx.x;
  int lane = tid & 31;
  int wave = tid >> 5;                  // 0..15
  int dir  = blockIdx.x >> 2;
  int b0   = (blockIdx.x & 3) * 16;
  const float*  gxd = gx + (size_t)dir * BT_ * G4_;
  const __bf16* wp  = whh + (size_t)dir * (64 * 8 * 32 * 16);
  for (int i = tid; i < 16 * H_; i += 512) hbuf[i] = (__bf16)0.0f;
  float cst[8];
  for (int e = 0; e < 8; ++e) cst[e] = 0.f;
  int mrow = (lane >> 4) << 3;               // C-fragment row base: 0 or 8
  int jcol = wave * 16 + (lane & 15);        // hidden column
  __syncthreads();
  for (int tt = 0; tt < T_; ++tt) {
    int t = dir ? (T_ - 1 - tt) : tt;
    // C init = precomputed input gates gx[:, t, :]
    v8f acc[4];
    for (int q = 0; q < 4; ++q) {
      int col = (q * 16 + wave) * 16 + (lane & 15);
      for (int e = 0; e < 8; ++e)
        acc[q][e] = gxd[((size_t)(b0 + mrow + e) * T_ + t) * G4_ + col];
    }
    // A fragments (h_prev) from LDS
    v16bf af[8];
    const __bf16* hrow = &hbuf[(lane & 15) * H_ + ((lane >> 4) << 3)];
    for (int kc = 0; kc < 8; ++kc) af[kc] = ld_fragA(hrow + kc * 32);
    float hp[8];
    for (int e = 0; e < 8; ++e)
      hp[e] = (float)hbuf[(mrow + e) * H_ + jcol];
    __syncthreads();                         // all hbuf reads complete
    // gates += h_prev @ W_hh^T   (B fragments stream from WGP$/L2)
    for (int q = 0; q < 4; ++q) {
      int nt = q * 16 + wave;
      for (int kc = 0; kc < 8; ++kc) {
        v16bf b = ld_fragP(wp + ((size_t)(nt * 8 + kc) * 32 + lane) * 16);
        acc[q] = __builtin_amdgcn_wmma_f32_16x16x32_bf16(
            false, af[kc], false, b, (short)0, acc[q], false, false);
      }
    }
    // pointwise LSTM cell + mask, write h to LDS + outputs
    for (int e = 0; e < 8; ++e) {
      int br = b0 + mrow + e;
      float ig = sigmoidf_(acc[0][e]);
      float fg = sigmoidf_(acc[1][e]);
      float gg = tanhf(acc[2][e]);
      float og = sigmoidf_(acc[3][e]);
      float cn = fg * cst[e] + ig * gg;
      float hn = og * tanhf(cn);
      float m  = mask[br * T_ + t];
      float hns = m * hn + (1.f - m) * hp[e];
      cst[e]    = m * cn + (1.f - m) * cst[e];
      hbuf[(mrow + e) * H_ + jcol] = (__bf16)hns;
      size_t o = ((size_t)br * T_ + t) * DP_ + dir * H_ + jcol;
      if (outf) outf[o] = hns;
      outb[o] = (__bf16)hns;
    }
    __syncthreads();                         // writes visible before next step
  }
}

// ---------------------------------------------------------------------------
extern "C" void kernel_launch(void* const* d_in, const int* in_sizes, int n_in,
                              void* d_out, int out_size, void* d_ws, size_t ws_size,
                              hipStream_t stream) {
  const int*   iw       = (const int*)d_in[0];
  const int*   ic       = (const int*)d_in[1];
  const int*   ip       = (const int*)d_in[2];
  const float* mask     = (const float*)d_in[3];
  const float* emb_word = (const float*)d_in[4];
  const float* emb_char = (const float*)d_in[5];
  const float* emb_pos  = (const float*)d_in[6];
  const float* conv_w   = (const float*)d_in[7];
  const float* conv_b   = (const float*)d_in[8];
  const float* wih[3]  = {(const float*)d_in[9],  (const float*)d_in[12], (const float*)d_in[15]};
  const float* whh[3]  = {(const float*)d_in[10], (const float*)d_in[13], (const float*)d_in[16]};
  const float* bias[3] = {(const float*)d_in[11], (const float*)d_in[14], (const float*)d_in[17]};
  const int Kreal[3] = {500, 512, 512};

  char* ws = (char*)d_ws;
  const size_t MB = 1ull << 20;
  float*  S   = (float*)(ws + 0);
  __bf16* WIH = (__bf16*)(ws + 1 * MB);    // 6 x 1MB
  __bf16* WHH = (__bf16*)(ws + 7 * MB);    // 6 x 0.5MB
  __bf16* XA  = (__bf16*)(ws + 16 * MB);   // 16MB
  __bf16* XB  = (__bf16*)(ws + 32 * MB);   // 16MB
  float*  GX  = (float*)(ws + 48 * MB);    // 128MB
  const size_t WIH_SZ = 64 * 16 * 32 * 16; // bf16 elems per layer-dir
  const size_t WHH_SZ = 64 * 8 * 32 * 16;

  for (int l = 0; l < 3; ++l)
    for (int d = 0; d < 2; ++d) {
      pack_w_kernel<<<64 * 16, 32, 0, stream>>>(
          wih[l] + (size_t)d * G4_ * Kreal[l], Kreal[l], 16,
          WIH + (size_t)(2 * l + d) * WIH_SZ);
      pack_w_kernel<<<64 * 8, 32, 0, stream>>>(
          whh[l] + (size_t)d * G4_ * H_, H_, 8,
          WHH + (size_t)(2 * l + d) * WHH_SZ);
    }
  build_S_kernel<<<((CV_ + 1) * KC_ * NF_ + 255) / 256, 256, 0, stream>>>(
      emb_char, conv_w, S);
  embed_kernel<<<BT_ / TOK_PER_BLK, 128, 0, stream>>>(
      iw, ic, ip, emb_word, emb_pos, S, conv_b, XA);

  __bf16* cur = XA;
  __bf16* nxt = XB;
  for (int l = 0; l < 3; ++l) {
    gemm_gx_kernel<<<16384, 256, 0, stream>>>(
        cur, WIH + (size_t)(2 * l) * WIH_SZ, bias[l], GX);
    float* of = (l == 2) ? (float*)d_out : nullptr;
    lstm_scan_kernel<<<8, 512, 0, stream>>>(
        GX, WHH + (size_t)(2 * l) * WHH_SZ, mask, of, nxt);
    __bf16* tmp = cur; cur = nxt; nxt = tmp;
  }
}
